// SpatialAxialAttention_34926674051602
// MI455X (gfx1250) — matrix operand order repaired and checked by reading, not compiled, not run on hardware
//
#include <hip/hip_runtime.h>
#include <hip/hip_bf16.h>

// ---------------------------------------------------------------------------
// SpatialAxialAttention for MI455X (gfx1250, wave32, WMMA bf16 16x16x32 + TDM)
// ---------------------------------------------------------------------------
// Shapes (fixed by reference):
//   x      : [8, 1024, 1024]  (B*T=8, H*W=1024 seq, D=1024) fp32
//   w_qkv  : [1024, 3072] fp32     w_out : [1024, 1024] fp32   b_out : [1024]
//   HEADS=16, head_dim=64, scale = 1/8
// ---------------------------------------------------------------------------

typedef __bf16 bf16;
typedef __attribute__((ext_vector_type(8)))  __bf16 bf16x8;
typedef __attribute__((ext_vector_type(16))) __bf16 bf16x16;
typedef __attribute__((ext_vector_type(8)))  float  f32x8;
typedef __attribute__((ext_vector_type(4)))  unsigned int u32x4;
typedef __attribute__((ext_vector_type(8)))  int i32x8;
typedef __attribute__((ext_vector_type(4)))  int i32x4;

#if defined(__has_builtin)
#if __has_builtin(__builtin_amdgcn_tensor_load_to_lds)
#define HAS_TDM 1
#endif
#endif

#define WMMA_BF16(a, b, c) \
  __builtin_amdgcn_wmma_f32_16x16x32_bf16(false, (a), false, (b), (short)0, (c), false, false)

static __device__ __forceinline__ bf16x16 cat8(bf16x8 lo, bf16x8 hi) {
  return __builtin_shufflevector(lo, hi, 0,1,2,3,4,5,6,7,8,9,10,11,12,13,14,15);
}
static __device__ __forceinline__ f32x8 zero8() {
  f32x8 z;
  #pragma unroll
  for (int i = 0; i < 8; ++i) z[i] = 0.0f;
  return z;
}

#ifdef HAS_TDM
// TDM 2-D tile load: tile = 128 rows x 32 bf16, row pitch in LDS padded
// 64B + 16B (pad_interval = 16 DWORDs, pad_amount = 4 DWORDs) -> 40-element pitch.
// This toolchain exposes the 6-arg builtin:
//   (uint32x4 g0, int32x8 g1, int32x4 g2, int32x4 g3, int32x8 g4, int cpol)
static __device__ __forceinline__ void tdm_load_tile(unsigned int lds_addr,
                                                     unsigned long long gaddr,
                                                     unsigned int dim0,
                                                     unsigned int dim1,
                                                     unsigned int stride0) {
  // D# group0 (128b): count=1 | lds_addr | global_addr[56:0] | type=2
  u32x4 g0 = { 1u, lds_addr, (unsigned int)gaddr,
               (unsigned int)(gaddr >> 32) | (2u << 30) };
  // D# group1 (256b)
  const unsigned int w0 = (1u << 16)   // data_size = 1 (2 bytes)
                        | (1u << 20)   // pad_enable
                        | (3u << 22)   // pad_interval: 16 DWORDs (64B) before pad
                        | (3u << 25);  // pad_amount: 4 DWORDs (16B)
  i32x8 g1 = { (int)w0,
               (int)((dim0 & 0xFFFFu) << 16),                    // [31:16]=tensor_dim0.lo
               (int)((dim0 >> 16) | ((dim1 & 0xFFFFu) << 16)),   // dim0.hi | dim1.lo
               (int)((dim1 >> 16) | (32u << 16)),                // dim1.hi | tile_dim0=32
               (int)128u,                                        // tile_dim1=128, tile_dim2=0
               (int)stride0,                                     // tensor_dim0_stride[31:0]
               0, 0 };
  i32x4 z4 = { 0, 0, 0, 0 };
  i32x8 z8 = { 0, 0, 0, 0, 0, 0, 0, 0 };
  __builtin_amdgcn_tensor_load_to_lds(g0, g1, z4, z4, z8, 0);
}
#endif

// ---------------------------------------------------------------------------
// fp32 -> bf16 convert
__global__ void k_f32_to_bf16(const float* __restrict__ src, bf16* __restrict__ dst, int n) {
  int i = blockIdx.x * blockDim.x + threadIdx.x;
  if (i < n) dst[i] = (bf16)src[i];
}

// fp32 [K][N] -> bf16 [N][K] (transpose so GEMM B operand is row-per-N)
__global__ void k_transpose_bf16(const float* __restrict__ src, bf16* __restrict__ dst,
                                 int K, int N) {
  int i = blockIdx.x * blockDim.x + threadIdx.x;
  if (i < K * N) {
    int k = i / N, n = i - k * N;
    dst[(size_t)n * K + k] = (bf16)src[i];
  }
}

// ---------------------------------------------------------------------------
// Tiled bf16 WMMA GEMM:  C[M,N] = A[M,K] * BT[N,K]^T
// Block = 256 threads (8 waves), tile 128x128, K-step 32.
// TDM double-buffered pipeline: wave 0 DMAs next K-step's tiles into the
// alternate LDS buffer, waits TENSORcnt<=2 (in-order => current buffer done),
// barrier fans completion out to the other waves.
// EPI==0: store bf16 into Cb ; EPI==1: store fp32 + bias into Cf.
template <int EPI>
__global__ __launch_bounds__(256) void k_gemm(const bf16* __restrict__ A,
                                              const bf16* __restrict__ BT,
                                              int M, int K, int N,
                                              bf16* __restrict__ Cb,
                                              float* __restrict__ Cf,
                                              const float* __restrict__ bias) {
  __shared__ bf16 sA[2][128 * 40];  // 32 cols + 8 pad per row
  __shared__ bf16 sB[2][128 * 40];

  const int tid  = threadIdx.x;
  const int l    = tid & 31;
  const int wv   = tid >> 5;
  const int l16  = l & 15;
  const int half = l >> 4;
  const int wm   = wv & 1;   // 0..1 : 64-row half
  const int wn   = wv >> 1;  // 0..3 : 32-col quarter
  const int rowBase = blockIdx.x * 128;
  const int colBase = blockIdx.y * 128;

  f32x8 acc[4][2];
  #pragma unroll
  for (int i = 0; i < 4; ++i)
    #pragma unroll
    for (int j = 0; j < 2; ++j) acc[i][j] = zero8();

#ifdef HAS_TDM
  const unsigned long long aG = (unsigned long long)(uintptr_t)(A + (size_t)rowBase * K);
  const unsigned long long bG = (unsigned long long)(uintptr_t)(BT + (size_t)colBase * K);
  const unsigned int ldsA0 = (unsigned int)(uintptr_t)&sA[0][0];
  const unsigned int ldsA1 = (unsigned int)(uintptr_t)&sA[1][0];
  const unsigned int ldsB0 = (unsigned int)(uintptr_t)&sB[0][0];
  const unsigned int ldsB1 = (unsigned int)(uintptr_t)&sB[1][0];
  if (wv == 0) {  // prime the pipeline: buffer 0 <- k0 = 0
    tdm_load_tile(ldsA0, aG, (unsigned)K, (unsigned)M, (unsigned)K);
    tdm_load_tile(ldsB0, bG, (unsigned)K, (unsigned)N, (unsigned)K);
  }
#endif

  for (int k0 = 0; k0 < K; k0 += 32) {
    const int cur = (k0 >> 5) & 1;
    __syncthreads();  // all waves done with buffer cur^1 before it is refilled
#ifdef HAS_TDM
    if (wv == 0) {
      if (k0 + 32 < K) {  // prefetch next K-step into the other buffer
        tdm_load_tile(cur ? ldsA0 : ldsA1, aG + (unsigned long long)(k0 + 32) * 2,
                      (unsigned)K, (unsigned)M, (unsigned)K);
        tdm_load_tile(cur ? ldsB0 : ldsB1, bG + (unsigned long long)(k0 + 32) * 2,
                      (unsigned)K, (unsigned)N, (unsigned)K);
        __builtin_amdgcn_s_wait_tensorcnt(2);  // oldest 2 (current buffer) complete
      } else {
        __builtin_amdgcn_s_wait_tensorcnt(0);
      }
    }
#else
    // fallback: cooperative tile load, 512 16-byte chunks each for A and B
    #pragma unroll
    for (int c = tid; c < 512; c += 256) {
      int r  = c >> 2;
      int cc = (c & 3) * 8;
      *(bf16x8*)(&sA[cur][r * 40 + cc]) =
          *(const bf16x8*)(&A[(size_t)(rowBase + r) * K + k0 + cc]);
      *(bf16x8*)(&sB[cur][r * 40 + cc]) =
          *(const bf16x8*)(&BT[(size_t)(colBase + r) * K + k0 + cc]);
    }
#endif
    __syncthreads();

    bf16x16 af[4], bfr[2];
    #pragma unroll
    for (int i = 0; i < 4; ++i) {
      // A frag layout: lanes 0-15 K{0..7,16..23}, lanes 16-31 K{8..15,24..31}
      const bf16* base = &sA[cur][(wm * 64 + i * 16 + l16) * 40];
      bf16x8 lo = *(const bf16x8*)(base + half * 8);
      bf16x8 hi = *(const bf16x8*)(base + 16 + half * 8);
      af[i] = cat8(lo, hi);
    }
    #pragma unroll
    for (int j = 0; j < 2; ++j) {
      // B frag layout: lanes 0-15 K{0..7,16..23}, lanes 16-31 K{8..15,24..31}, N = lane%16
      const bf16* base = &sB[cur][(wn * 32 + j * 16 + l16) * 40 + half * 8];
      bf16x8 lo = *(const bf16x8*)(base);
      bf16x8 hi = *(const bf16x8*)(base + 16);
      bfr[j] = cat8(lo, hi);
    }
    #pragma unroll
    for (int i = 0; i < 4; ++i)
      #pragma unroll
      for (int j = 0; j < 2; ++j)
        acc[i][j] = WMMA_BF16(af[i], bfr[j], acc[i][j]);
  }

  // epilogue: C frag = vgpr r -> row r (+8 for upper half-wave), col = lane%16
  #pragma unroll
  for (int i = 0; i < 4; ++i) {
    #pragma unroll
    for (int j = 0; j < 2; ++j) {
      #pragma unroll
      for (int r = 0; r < 8; ++r) {
        int gr = rowBase + wm * 64 + i * 16 + r + half * 8;
        int gc = colBase + wn * 32 + j * 16 + l16;
        float v = acc[i][j][r];
        if (EPI == 0) Cb[(size_t)gr * N + gc] = (bf16)v;
        else          Cf[(size_t)gr * N + gc] = v + bias[gc];
      }
    }
  }
}

// ---------------------------------------------------------------------------
// RoPE (pixel freqs) + repack: qkv bf16 [8192][3072] ->
//   qh,kh : [bh=128][seq=1024][64] (rotated), vT : [bh=128][64][1024]
__global__ void k_rope_pack(const bf16* __restrict__ qkv,
                            bf16* __restrict__ qh, bf16* __restrict__ kh,
                            bf16* __restrict__ vT) {
  int t    = blockIdx.x * 256 + threadIdx.x;   // 8*16*1024*32 = 4194304 threads
  int d2   = t & 31;                           // dim pair index 0..31
  int seq  = (t >> 5) & 1023;
  int head = (t >> 15) & 15;
  int bt   = t >> 19;
  int bh   = bt * 16 + head;

  size_t row = (size_t)(bt * 1024 + seq);
  int    col = head * 64 + d2 * 2;
  const bf16* rp = qkv + row * 3072;
  float q0 = (float)rp[col],        q1 = (float)rp[col + 1];
  float k0 = (float)rp[1024 + col], k1 = (float)rp[1024 + col + 1];
  float v0 = (float)rp[2048 + col], v1 = (float)rp[2048 + col + 1];

  // freqs_for='pixel': base = linspace(1, 128, 16) * pi ; pos in [-1, 1]
  int   j    = d2 & 15;
  float base = (1.0f + (127.0f / 15.0f) * (float)j) * 3.14159265358979323f;
  float pos  = (d2 < 16) ? (-1.0f + (2.0f / 31.0f) * (float)(seq >> 5))   // h axis
                         : (-1.0f + (2.0f / 31.0f) * (float)(seq & 31));  // w axis
  float f = pos * base, s, c;
  __sincosf(f, &s, &c);

  size_t o = (size_t)bh * 65536 + (size_t)seq * 64 + d2 * 2;
  qh[o]     = (bf16)(q0 * c - q1 * s);
  qh[o + 1] = (bf16)(q1 * c + q0 * s);
  kh[o]     = (bf16)(k0 * c - k1 * s);
  kh[o + 1] = (bf16)(k1 * c + k0 * s);

  size_t ov = (size_t)bh * 65536 + (size_t)(d2 * 2) * 1024 + seq;
  vT[ov]        = (bf16)v0;
  vT[ov + 1024] = (bf16)v1;
}

// ---------------------------------------------------------------------------
// Flash attention: 1 wave = 16 Q rows, online softmax, 32 keys per step.
// Output written bf16 into attn [8192][1024] (row = bt*1024+seq, col = head*64+d)
__global__ __launch_bounds__(256) void k_attention(const bf16* __restrict__ qh,
                                                   const bf16* __restrict__ kh,
                                                   const bf16* __restrict__ vT,
                                                   bf16* __restrict__ attn) {
  __shared__ bf16 pbuf[8][16 * 32];  // per-wave P staging (C-layout -> A-layout)

  const int tid  = threadIdx.x;
  const int wv   = tid >> 5;
  const int l    = tid & 31;
  const int l16  = l & 15;
  const int half = l >> 4;
  const int bh   = blockIdx.x >> 3;                       // 0..127
  const int q0r  = (blockIdx.x & 7) * 128 + wv * 16;      // Q tile base row

  const bf16* Q  = qh + (size_t)bh * 65536;
  const bf16* Kp = kh + (size_t)bh * 65536;
  const bf16* Vp = vT + (size_t)bh * 65536;

  // Q A-fragments for dim chunks [0..31], [32..63]
  bf16x16 qa[2];
  #pragma unroll
  for (int ci = 0; ci < 2; ++ci) {
    const bf16* base = Q + (size_t)(q0r + l16) * 64 + ci * 32 + half * 8;
    qa[ci] = cat8(*(const bf16x8*)base, *(const bf16x8*)(base + 16));
  }

  f32x8 O[4];
  #pragma unroll
  for (int t = 0; t < 4; ++t) O[t] = zero8();
  float m[8], lsum[8];
  #pragma unroll
  for (int r = 0; r < 8; ++r) { m[r] = -3.0e38f; lsum[r] = 0.0f; }

  for (int j0 = 0; j0 < 1024; j0 += 32) {
    if (j0 + 32 < 1024) {  // prefetch next K/V tiles (global_prefetch_b8)
      __builtin_prefetch(Kp + (size_t)(j0 + 32 + l16) * 64, 0, 1);
      __builtin_prefetch(Vp + (size_t)l16 * 1024 + j0 + 32, 0, 1);
    }
    // S = Q * K^T for keys [j0..j0+15] (S0) and [j0+16..j0+31] (S1)
    f32x8 S0 = zero8(), S1 = zero8();
    #pragma unroll
    for (int ci = 0; ci < 2; ++ci) {
      {
        const bf16* base = Kp + (size_t)(j0 + l16) * 64 + ci * 32 + half * 8;
        S0 = WMMA_BF16(qa[ci], cat8(*(const bf16x8*)base, *(const bf16x8*)(base + 16)), S0);
      }
      {
        const bf16* base = Kp + (size_t)(j0 + 16 + l16) * 64 + ci * 32 + half * 8;
        S1 = WMMA_BF16(qa[ci], cat8(*(const bf16x8*)base, *(const bf16x8*)(base + 16)), S1);
      }
    }
    // online softmax (rows live across the 16-lane N-groups of the C layout)
    float p0[8], p1[8];
    #pragma unroll
    for (int r = 0; r < 8; ++r) {
      float s0 = S0[r] * 0.125f, s1 = S1[r] * 0.125f;
      float rowmax = fmaxf(s0, s1);
      #pragma unroll
      for (int off = 1; off < 16; off <<= 1)
        rowmax = fmaxf(rowmax, __shfl_xor(rowmax, off, 32));
      float mn    = fmaxf(m[r], rowmax);
      float alpha = __expf(m[r] - mn);
      m[r] = mn;
      float e0 = __expf(s0 - mn), e1 = __expf(s1 - mn);
      float rs = e0 + e1;
      #pragma unroll
      for (int off = 1; off < 16; off <<= 1) rs += __shfl_xor(rs, off, 32);
      lsum[r] = lsum[r] * alpha + rs;
      p0[r] = e0; p1[r] = e1;
      #pragma unroll
      for (int t = 0; t < 4; ++t) O[t][r] *= alpha;
    }
    // stage P (C layout) to LDS, reload as A fragment
    #pragma unroll
    for (int r = 0; r < 8; ++r) {
      int prow = r + half * 8;
      pbuf[wv][prow * 32 + l16]      = (bf16)p0[r];
      pbuf[wv][prow * 32 + 16 + l16] = (bf16)p1[r];
    }
    asm volatile("s_wait_dscnt 0" ::: "memory");  // DS in-order per wave; fence compiler too
    bf16x16 pa;
    {
      const bf16* base = &pbuf[wv][l16 * 32 + half * 8];
      pa = cat8(*(const bf16x8*)base, *(const bf16x8*)(base + 16));
    }
    // O += P * V   (V pre-transposed -> contiguous B fragments)
    #pragma unroll
    for (int t = 0; t < 4; ++t) {
      const bf16* base = Vp + (size_t)(t * 16 + l16) * 1024 + j0 + half * 8;
      O[t] = WMMA_BF16(pa, cat8(*(const bf16x8*)base, *(const bf16x8*)(base + 16)), O[t]);
    }
  }

  const int bt = bh >> 4, head = bh & 15;
  #pragma unroll
  for (int t = 0; t < 4; ++t)
    #pragma unroll
    for (int r = 0; r < 8; ++r) {
      int grow = bt * 1024 + q0r + r + half * 8;
      int gcol = head * 64 + t * 16 + l16;
      attn[(size_t)grow * 1024 + gcol] = (bf16)(O[t][r] / lsum[r]);
    }
}

// ---------------------------------------------------------------------------
extern "C" void kernel_launch(void* const* d_in, const int* in_sizes, int n_in,
                              void* d_out, int out_size, void* d_ws, size_t ws_size,
                              hipStream_t stream) {
  const float* x     = (const float*)d_in[0];  // [8,1024,1024]
  const float* w_qkv = (const float*)d_in[1];  // [1024,3072]
  const float* w_out = (const float*)d_in[2];  // [1024,1024]
  const float* b_out = (const float*)d_in[3];  // [1024]
  float*       out   = (float*)d_out;          // [8192,1024] fp32

  char* ws = (char*)d_ws;
  size_t off = 0;
  bf16* xb    = (bf16*)(ws + off); off += (size_t)8192 * 1024 * 2;   // 16 MB
  bf16* wqkvT = (bf16*)(ws + off); off += (size_t)3072 * 1024 * 2;   //  6 MB
  bf16* woutT = (bf16*)(ws + off); off += (size_t)1024 * 1024 * 2;   //  2 MB
  bf16* qkv   = (bf16*)(ws + off); off += (size_t)8192 * 3072 * 2;   // 48 MB
  bf16* qhp   = (bf16*)(ws + off); off += (size_t)128 * 65536 * 2;   // 16 MB
  bf16* khp   = (bf16*)(ws + off); off += (size_t)128 * 65536 * 2;   // 16 MB
  bf16* vTp   = (bf16*)(ws + off); off += (size_t)128 * 65536 * 2;   // 16 MB
  bf16* attn  = (bf16*)(ws + off); off += (size_t)8192 * 1024 * 2;   // 16 MB

  // 1) convert inputs to bf16 (weights transposed to [N][K])
  k_f32_to_bf16<<<8388608 / 256, 256, 0, stream>>>(x, xb, 8388608);
  k_transpose_bf16<<<(3072 * 1024) / 256, 256, 0, stream>>>(w_qkv, wqkvT, 1024, 3072);
  k_transpose_bf16<<<(1024 * 1024) / 256, 256, 0, stream>>>(w_out, woutT, 1024, 1024);

  // 2) QKV projection: [8192,1024] x [1024,3072] -> bf16 qkv  (TDM pipelined)
  k_gemm<0><<<dim3(64, 24), 256, 0, stream>>>(xb, wqkvT, 8192, 1024, 3072,
                                              qkv, nullptr, nullptr);

  // 3) RoPE + head-major repack (V transposed)
  k_rope_pack<<<16384, 256, 0, stream>>>(qkv, qhp, khp, vTp);

  // 4) flash attention (128 bh-problems, 1024 seq, hd 64)
  k_attention<<<1024, 256, 0, stream>>>(qhp, khp, vTp, attn);

  // 5) output projection + bias: [8192,1024] x [1024,1024] -> fp32 out
  k_gemm<1><<<dim3(64, 8), 256, 0, stream>>>(attn, woutT, 8192, 1024, 1024,
                                             nullptr, out, b_out);
}
